// EraseAddWriteOperation_62929860821077
// MI455X (gfx1250) — matrix operand order repaired and checked by reading, not compiled, not run on hardware
//
#include <hip/hip_runtime.h>
#include <hip/hip_bf16.h>

// ---------------------------------------------------------------------------
// EraseAddWriteOperation for MI455X (gfx1250, wave32, WMMA bf16).
// B=8, M=8192, D=512, R=64, H=128.
// ---------------------------------------------------------------------------

#define DD 512
#define HH 128
#define RR 64
#define MM 8192
#define BB 8

typedef __attribute__((ext_vector_type(16))) __bf16       v16bf;
typedef __attribute__((ext_vector_type(8)))  float        v8f;
typedef __attribute__((ext_vector_type(8)))  unsigned int v8u;

__device__ __forceinline__ unsigned short f2bf(float f) {
    unsigned int u = __builtin_bit_cast(unsigned int, f);
    u += 0x7fffu + ((u >> 16) & 1u);     // round-to-nearest-even
    return (unsigned short)(u >> 16);
}

__device__ __forceinline__ float gelu_tanh(float x) {
    float x3 = x * x * x;
    float t  = tanhf(0.7978845608028654f * (x + 0.044715f * x3));
    return 0.5f * x * (1.0f + t);
}

// A-fragment K offset for 16-bit WMMA A (16x32), chunk-local, contiguous k = 16*g + 2*v (matches pk2 B fills).
__device__ __forceinline__ int afrag_k(int v, int g) {
    return g * 16 + v * 2;   // g*16 + 2*v
}

// ---------------------------------------------------------------------------
// Repack f32 [K][N] weights to bf16 "pk2": dst[(k>>1)*(2N) + n*2 + (k&1)].
// A 32-bit read at uint index (k/2)*N + n then yields the (k,k+1) bf16 pair
// exactly as one B-fragment VGPR expects.
// ---------------------------------------------------------------------------
__global__ __launch_bounds__(256) void pack_pk2(const float* __restrict__ src,
                                                unsigned short* __restrict__ dst,
                                                int K, int N) {
    int i = blockIdx.x * 256 + threadIdx.x;
    if (i >= K * N) return;
    int k = i / N, n = i - k * N;
    dst[(k >> 1) * (N * 2) + n * 2 + (k & 1)] = f2bf(src[i]);
}

// ---------------------------------------------------------------------------
// Erase path: LN -> Dense(H) -> gelu -> Dense(R). One wave = 16 tokens.
// elog layout: [b][m][R] (f32, coalesced stores).
// ---------------------------------------------------------------------------
__global__ __launch_bounds__(128) void erase_logits_kernel(
    const float* __restrict__ mem,
    const float* __restrict__ ln_s, const float* __restrict__ ln_b,
    const unsigned int* __restrict__ W1p, const float* __restrict__ b1,
    const unsigned int* __restrict__ W2p, const float* __restrict__ b2,
    float* __restrict__ elog) {
    __shared__ unsigned short lds[4][16 * DD];          // 64 KB / block
    const int lane = threadIdx.x & 31;
    const int w    = threadIdx.x >> 5;
    const int g    = lane >> 4;
    const int hn   = lane & 15;
    const long long tok0 = ((long long)blockIdx.x * 4 + w) * 16;
    unsigned short* A = lds[w];

    // ---- LayerNorm 16 tokens -> bf16 tile [16][512] in LDS ----
    for (int t = 0; t < 16; ++t) {
        const float* xp = mem + (tok0 + t) * DD;
        float xv[16];
        float s = 0.f, ss = 0.f;
#pragma unroll
        for (int i = 0; i < 16; ++i) {
            float v = xp[i * 32 + lane];
            xv[i] = v; s += v; ss += v * v;
        }
#pragma unroll
        for (int o = 16; o > 0; o >>= 1) {
            s  += __shfl_xor(s,  o, 32);
            ss += __shfl_xor(ss, o, 32);
        }
        float mean = s * (1.0f / DD);
        float var  = ss * (1.0f / DD) - mean * mean;
        float rstd = rsqrtf(var + 1e-6f);
#pragma unroll
        for (int i = 0; i < 16; ++i) {
            int k = i * 32 + lane;
            A[t * DD + k] = f2bf((xv[i] - mean) * rstd * ln_s[k] + ln_b[k]);
        }
    }
    __syncthreads();

    // ---- GEMM1: [16x512] @ [512x128] -> 8 accumulator tiles ----
    v8f vzero = {};
    v8f acc[8];
#pragma unroll
    for (int n = 0; n < 8; ++n) acc[n] = vzero;
#pragma unroll 4
    for (int c = 0; c < 16; ++c) {
        int k0 = c * 32;
        v8u au;
#pragma unroll
        for (int v = 0; v < 8; ++v)
            au[v] = *(const unsigned int*)&A[hn * DD + k0 + afrag_k(v, g)];
        v16bf af = __builtin_bit_cast(v16bf, au);
        int pbase = (k0 >> 1) + g * 8;
#pragma unroll
        for (int n = 0; n < 8; ++n) {
            v8u bu;
#pragma unroll
            for (int v = 0; v < 8; ++v)
                bu[v] = W1p[(pbase + v) * HH + n * 16 + hn];
            v16bf bfv = __builtin_bit_cast(v16bf, bu);
            acc[n] = __builtin_amdgcn_wmma_f32_16x16x32_bf16(
                false, af, false, bfv, (short)0, acc[n], false, false);
        }
    }
    __syncthreads();

    // ---- bias + gelu -> bf16 h tile [16][128] back into LDS ----
#pragma unroll
    for (int n = 0; n < 8; ++n) {
        float bias = b1[n * 16 + hn];
#pragma unroll
        for (int j = 0; j < 8; ++j) {
            int m = j + 8 * g;                       // C/D layout: lanes16-31 -> M+8
            A[m * HH + n * 16 + hn] = f2bf(gelu_tanh(acc[n][j] + bias));
        }
    }
    __syncthreads();

    // ---- GEMM2: [16x128] @ [128x64] ----
    v8f acc2[4];
#pragma unroll
    for (int n = 0; n < 4; ++n) acc2[n] = vzero;
#pragma unroll
    for (int c = 0; c < 4; ++c) {
        int k0 = c * 32;
        v8u au;
#pragma unroll
        for (int v = 0; v < 8; ++v)
            au[v] = *(const unsigned int*)&A[hn * HH + k0 + afrag_k(v, g)];
        v16bf af = __builtin_bit_cast(v16bf, au);
        int pbase = (k0 >> 1) + g * 8;
#pragma unroll
        for (int n = 0; n < 4; ++n) {
            v8u bu;
#pragma unroll
            for (int v = 0; v < 8; ++v)
                bu[v] = W2p[(pbase + v) * RR + n * 16 + hn];
            v16bf bfv = __builtin_bit_cast(v16bf, bu);
            acc2[n] = __builtin_amdgcn_wmma_f32_16x16x32_bf16(
                false, af, false, bfv, (short)0, acc2[n], false, false);
        }
    }
#pragma unroll
    for (int n = 0; n < 4; ++n) {
        float bias = b2[n * 16 + hn];
#pragma unroll
        for (int j = 0; j < 8; ++j) {
            int m = j + 8 * g;
            elog[(tok0 + m) * RR + n * 16 + hn] = acc2[n][j] + bias;
        }
    }
}

// ---------------------------------------------------------------------------
// Softmax over m for each (b,r). Reads f32 elog[b][m][R], writes bf16
// esm[b][m][R] (which is precisely the pk2 A-matrix layout for the einsum).
// ---------------------------------------------------------------------------
__global__ __launch_bounds__(256) void softmax_kernel(
    const float* __restrict__ elog, unsigned short* __restrict__ esm) {
    __shared__ float red[256];
    int b = blockIdx.x >> 6;
    int r = blockIdx.x & 63;
    const float* src = elog + (long long)b * MM * RR + r;
    unsigned short* dst = esm + (long long)b * MM * RR + r;
    int tid = threadIdx.x;
    float mx = -3.402823466e38f;
    for (int m = tid; m < MM; m += 256)
        mx = fmaxf(mx, src[(long long)m * RR]);
    red[tid] = mx; __syncthreads();
    for (int o = 128; o > 0; o >>= 1) {
        if (tid < o) red[tid] = fmaxf(red[tid], red[tid + o]);
        __syncthreads();
    }
    mx = red[0]; __syncthreads();
    float sum = 0.f;
    for (int m = tid; m < MM; m += 256)
        sum += __expf(src[(long long)m * RR] - mx);
    red[tid] = sum; __syncthreads();
    for (int o = 128; o > 0; o >>= 1) {
        if (tid < o) red[tid] += red[tid + o];
        __syncthreads();
    }
    float inv = 1.0f / red[0];
    for (int m = tid; m < MM; m += 256)
        dst[(long long)m * RR] = f2bf(__expf(src[(long long)m * RR] - mx) * inv);
}

// ---------------------------------------------------------------------------
// Add path: LN -> Dense(H) -> gelu -> Dense(D). One block = one batch
// (4 waves x 16 tokens = R=64). Output packed bf16 as the einsum B matrix:
// add_pk uint layout [b][r/2][d]  (low half = even r, high half = odd r).
// ---------------------------------------------------------------------------
__global__ __launch_bounds__(128) void add_path_kernel(
    const float* __restrict__ otok,
    const float* __restrict__ ln_s, const float* __restrict__ ln_b,
    const unsigned int* __restrict__ W1p, const float* __restrict__ b1,
    const unsigned int* __restrict__ W2p, const float* __restrict__ b2,
    unsigned short* __restrict__ add_pk) {
    __shared__ unsigned short lds[4][16 * DD];
    const int lane = threadIdx.x & 31;
    const int w    = threadIdx.x >> 5;
    const int g    = lane >> 4;
    const int hn   = lane & 15;
    const int b    = blockIdx.x;
    unsigned short* A = lds[w];

    for (int t = 0; t < 16; ++t) {
        const float* xp = otok + ((long long)b * RR + w * 16 + t) * DD;
        float xv[16];
        float s = 0.f, ss = 0.f;
#pragma unroll
        for (int i = 0; i < 16; ++i) {
            float v = xp[i * 32 + lane];
            xv[i] = v; s += v; ss += v * v;
        }
#pragma unroll
        for (int o = 16; o > 0; o >>= 1) {
            s  += __shfl_xor(s,  o, 32);
            ss += __shfl_xor(ss, o, 32);
        }
        float mean = s * (1.0f / DD);
        float rstd = rsqrtf(ss * (1.0f / DD) - mean * mean + 1e-6f);
#pragma unroll
        for (int i = 0; i < 16; ++i) {
            int k = i * 32 + lane;
            A[t * DD + k] = f2bf((xv[i] - mean) * rstd * ln_s[k] + ln_b[k]);
        }
    }
    __syncthreads();

    v8f vzero = {};
    v8f acc[8];
#pragma unroll
    for (int n = 0; n < 8; ++n) acc[n] = vzero;
#pragma unroll 4
    for (int c = 0; c < 16; ++c) {
        int k0 = c * 32;
        v8u au;
#pragma unroll
        for (int v = 0; v < 8; ++v)
            au[v] = *(const unsigned int*)&A[hn * DD + k0 + afrag_k(v, g)];
        v16bf af = __builtin_bit_cast(v16bf, au);
        int pbase = (k0 >> 1) + g * 8;
#pragma unroll
        for (int n = 0; n < 8; ++n) {
            v8u bu;
#pragma unroll
            for (int v = 0; v < 8; ++v)
                bu[v] = W1p[(pbase + v) * HH + n * 16 + hn];
            v16bf bfv = __builtin_bit_cast(v16bf, bu);
            acc[n] = __builtin_amdgcn_wmma_f32_16x16x32_bf16(
                false, af, false, bfv, (short)0, acc[n], false, false);
        }
    }
    __syncthreads();
#pragma unroll
    for (int n = 0; n < 8; ++n) {
        float bias = b1[n * 16 + hn];
#pragma unroll
        for (int j = 0; j < 8; ++j) {
            int m = j + 8 * g;
            A[m * HH + n * 16 + hn] = f2bf(gelu_tanh(acc[n][j] + bias));
        }
    }
    __syncthreads();

    // Preload A' fragments (K=128 -> 4 chunks), then stream N=512 in 32 tiles.
    v16bf afr[4];
#pragma unroll
    for (int c = 0; c < 4; ++c) {
        v8u au;
#pragma unroll
        for (int v = 0; v < 8; ++v)
            au[v] = *(const unsigned int*)&A[hn * HH + c * 32 + afrag_k(v, g)];
        afr[c] = __builtin_bit_cast(v16bf, au);
    }
    for (int nt = 0; nt < 32; ++nt) {
        v8f acc2 = vzero;
#pragma unroll
        for (int c = 0; c < 4; ++c) {
            int pbase = c * 16 + g * 8;
            v8u bu;
#pragma unroll
            for (int v = 0; v < 8; ++v)
                bu[v] = W2p[(pbase + v) * DD + nt * 16 + hn];
            v16bf bfv = __builtin_bit_cast(v16bf, bu);
            acc2 = __builtin_amdgcn_wmma_f32_16x16x32_bf16(
                false, afr[c], false, bfv, (short)0, acc2, false, false);
        }
        float bias = b2[nt * 16 + hn];
#pragma unroll
        for (int j = 0; j < 8; ++j) {
            int r = w * 16 + j + 8 * g;
            int d = nt * 16 + hn;
            add_pk[((long long)b * 32 + (r >> 1)) * (DD * 2) + d * 2 + (r & 1)] =
                f2bf(acc2[j] + bias);
        }
    }
}

// ---------------------------------------------------------------------------
// ea = erase^T @ add per batch (M=8192, N=512, K=64) then
// out = mem*(1-ea)+ea. One wave owns a 16-wide d tile and 512 m rows.
// ---------------------------------------------------------------------------
__global__ __launch_bounds__(128) void einsum_update_kernel(
    const unsigned short* __restrict__ esm,   // bf16 [b][m][64] (A, pk2)
    const unsigned int* __restrict__ add_pk,  // uint [b][32][512] (B frags)
    const float* __restrict__ mem,
    float* __restrict__ out) {
    const int lane  = threadIdx.x & 31;
    const int w     = threadIdx.x >> 5;
    const int g     = lane >> 4;
    const int hn    = lane & 15;
    const int wid   = blockIdx.x * 4 + w;
    const int strip = wid & 15;          // 16 strips of 512 rows
    const int nt    = (wid >> 4) & 31;   // 32 d tiles
    const int b     = wid >> 9;          // 8 batches

    v16bf bfr[2];
#pragma unroll
    for (int c = 0; c < 2; ++c) {
        int pbase = c * 16 + g * 8;
        v8u bu;
#pragma unroll
        for (int v = 0; v < 8; ++v)
            bu[v] = add_pk[((long long)b * 32 + pbase + v) * DD + nt * 16 + hn];
        bfr[c] = __builtin_bit_cast(v16bf, bu);
    }

    v8f vzero = {};
    for (int mt = 0; mt < 32; ++mt) {
        long long m0 = (long long)strip * 512 + mt * 16;
        v8f acc = vzero;
#pragma unroll
        for (int c = 0; c < 2; ++c) {
            int k0 = c * 32;
            v8u au;
#pragma unroll
            for (int v = 0; v < 8; ++v)
                au[v] = *(const unsigned int*)
                    &esm[((long long)b * MM + m0 + hn) * RR + k0 + afrag_k(v, g)];
            v16bf af = __builtin_bit_cast(v16bf, au);
            acc = __builtin_amdgcn_wmma_f32_16x16x32_bf16(
                false, af, false, bfr[c], (short)0, acc, false, false);
        }
#pragma unroll
        for (int j = 0; j < 8; ++j) {
            long long m   = m0 + j + 8 * g;
            long long idx = ((long long)b * MM + m) * DD + nt * 16 + hn;
            float mv = mem[idx];
            float ea = acc[j];
            out[idx] = mv * (1.0f - ea) + ea;
        }
    }
}

// ---------------------------------------------------------------------------
extern "C" void kernel_launch(void* const* d_in, const int* in_sizes, int n_in,
                              void* d_out, int out_size, void* d_ws, size_t ws_size,
                              hipStream_t stream) {
    (void)in_sizes; (void)n_in; (void)out_size; (void)ws_size;
    const float* memory = (const float*)d_in[0];
    const float* otok   = (const float*)d_in[1];
    const float* ln_e_s = (const float*)d_in[2];
    const float* ln_e_b = (const float*)d_in[3];
    const float* We1    = (const float*)d_in[4];
    const float* be1    = (const float*)d_in[5];
    const float* We2    = (const float*)d_in[6];
    const float* be2    = (const float*)d_in[7];
    const float* ln_a_s = (const float*)d_in[8];
    const float* ln_a_b = (const float*)d_in[9];
    const float* Wa1    = (const float*)d_in[10];
    const float* ba1    = (const float*)d_in[11];
    const float* Wa2    = (const float*)d_in[12];
    const float* ba2    = (const float*)d_in[13];
    float* out = (float*)d_out;

    char* ws = (char*)d_ws;
    unsigned short* We1p = (unsigned short*)ws;  ws += (size_t)DD * HH * 2;  // 128 KB
    unsigned short* We2p = (unsigned short*)ws;  ws += (size_t)HH * RR * 2;  // 16 KB
    unsigned short* Wa1p = (unsigned short*)ws;  ws += (size_t)DD * HH * 2;  // 128 KB
    unsigned short* Wa2p = (unsigned short*)ws;  ws += (size_t)HH * DD * 2;  // 128 KB
    float*          elog = (float*)ws;           ws += (size_t)BB * MM * RR * 4; // 16 MB
    unsigned short* esm  = (unsigned short*)ws;  ws += (size_t)BB * MM * RR * 2; // 8 MB
    unsigned short* addp = (unsigned short*)ws;  // 512 KB: [b][32][512] uints

    pack_pk2<<<(DD * HH + 255) / 256, 256, 0, stream>>>(We1, We1p, DD, HH);
    pack_pk2<<<(HH * RR + 255) / 256, 256, 0, stream>>>(We2, We2p, HH, RR);
    pack_pk2<<<(DD * HH + 255) / 256, 256, 0, stream>>>(Wa1, Wa1p, DD, HH);
    pack_pk2<<<(HH * DD + 255) / 256, 256, 0, stream>>>(Wa2, Wa2p, HH, DD);

    erase_logits_kernel<<<(BB * MM) / 64, 128, 0, stream>>>(
        memory, ln_e_s, ln_e_b,
        (const unsigned int*)We1p, be1, (const unsigned int*)We2p, be2, elog);

    add_path_kernel<<<BB, 128, 0, stream>>>(
        otok, ln_a_s, ln_a_b,
        (const unsigned int*)Wa1p, ba1, (const unsigned int*)Wa2p, ba2, addp);

    softmax_kernel<<<BB * RR, 256, 0, stream>>>(elog, esm);

    einsum_update_kernel<<<(BB * 32 * 16) / 4, 128, 0, stream>>>(
        esm, (const unsigned int*)addp, memory, out);
}
